// Attention_70300024701541
// MI455X (gfx1250) — matrix operand order repaired and checked by reading, not compiled, not run on hardware
//
#include <hip/hip_runtime.h>

// Problem constants (from reference): B=1, S=128, Q=256, C=256, H=8, DH=32.
#define S_   128
#define Q_   256
#define C_   256
#define H_   8
#define DH_  32
#define TOT_ 256
#define M_   (S_ * Q_)   // 32768 rows of [S*Q, C]

typedef __attribute__((ext_vector_type(16))) _Float16 v16h;
typedef __attribute__((ext_vector_type(8)))  _Float16 v8h;
typedef __attribute__((ext_vector_type(8)))  float    v8f;

static __device__ __forceinline__ v16h combine8(v8h lo, v8h hi) {
  return __builtin_shufflevector(lo, hi, 0,1,2,3,4,5,6,7,8,9,10,11,12,13,14,15);
}

// A fragment (16M x 32K, f16). `base` = row (lane&15) pointer already offset to
// column kbase. ISA layout: lanes 0-15 hold K=0..7 & 16..23, lanes 16-31 hold
// K=8..15 & 24..31 -> two contiguous 16B loads per lane.
static __device__ __forceinline__ v16h loadA(const _Float16* base, int lane) {
  const int kh = (lane >> 4) << 3;            // 0 or 8 halves
  v8h lo = *(const v8h*)(base + kh);
  v8h hi = *(const v8h*)(base + 16 + kh);
  return combine8(lo, hi);
}

// B fragment (32K x 16N, f16). `base` = pointer to "row n = lane&15" of a
// row-major [N][K] matrix at column kbase. Per lane (same as A): lanes 0-15 hold
// K=0..7 & 16..23, lanes 16-31 hold K=8..15 & 24..31 -> two contiguous 16B loads.
static __device__ __forceinline__ v16h loadB(const _Float16* base, int lane) {
  const _Float16* p = base + ((lane >> 4) << 3);
  v8h lo = *(const v8h*)(p);
  v8h hi = *(const v8h*)(p + 16);
  return combine8(lo, hi);
}

static __device__ __forceinline__ v8f wmma16x16x32(v16h a, v16h b, v8f c) {
  return __builtin_amdgcn_wmma_f32_16x16x32_f16(false, a, false, b,
                                                (short)0, c, false, false);
}

// xor-butterfly within each 16-lane row via v_permlane16_b32 (pure VALU; keeps
// the reduction off the DS pipe so it co-executes with WMMA/TRANS).
template <int MASK>
static __device__ __forceinline__ float permxor16(float v) {
  constexpr unsigned s0 =
      ((0u ^ MASK) << 0)  | ((1u ^ MASK) << 4)  | ((2u ^ MASK) << 8)  |
      ((3u ^ MASK) << 12) | ((4u ^ MASK) << 16) | ((5u ^ MASK) << 20) |
      ((6u ^ MASK) << 24) | ((7u ^ MASK) << 28);
  constexpr unsigned s1 =
      ((8u ^ MASK) << 0)  | ((9u ^ MASK) << 4)  | ((10u ^ MASK) << 8) |
      ((11u ^ MASK) << 12)| ((12u ^ MASK) << 16)| ((13u ^ MASK) << 20)|
      ((14u ^ MASK) << 24)| ((15u ^ MASK) << 28);
  unsigned u = __builtin_bit_cast(unsigned, v);
  unsigned r = __builtin_amdgcn_permlane16(u, u, s0, s1, false, false);
  return __builtin_bit_cast(float, r);
}

static __device__ __forceinline__ float half_reduce_max(float v) {
  v = fmaxf(v, permxor16<1>(v));
  v = fmaxf(v, permxor16<2>(v));
  v = fmaxf(v, permxor16<4>(v));
  v = fmaxf(v, permxor16<8>(v));
  return v;
}
static __device__ __forceinline__ float half_reduce_sum(float v) {
  v += permxor16<1>(v);
  v += permxor16<2>(v);
  v += permxor16<4>(v);
  v += permxor16<8>(v);
  return v;
}

// ---------------------------------------------------------------- convert ---
__global__ __launch_bounds__(256) void cvt_f32_f16(const float* __restrict__ src,
                                                   _Float16* __restrict__ dst,
                                                   int n) {
  int i = blockIdx.x * blockDim.x + threadIdx.x;
  const int stride = gridDim.x * blockDim.x;
  for (; i < n; i += stride) dst[i] = (_Float16)src[i];
}

// ------------------------------------------------------------- projections ---
// Each wave computes a 16(M) x 32(N) output tile: two WMMA N-tiles sharing one
// A fragment per K step.
// MODE 0: q -> [s,h,q,dh] f16   MODE 1: k -> [s,h,q,dh] f16
// MODE 2: v -> [s,h,dh,q] f16 (transposed for B-frag loads in attention)
// MODE 3: gate = sigmoid(y + bg) -> [m, 256] f32
template <int MODE>
__global__ __launch_bounds__(256) void proj_kernel(
    const _Float16* __restrict__ X,   // [M, C] f16
    const _Float16* __restrict__ W,   // [TOT, C] f16
    const float*   __restrict__ bg,   // [TOT] (MODE 3)
    _Float16* __restrict__ dst_h,
    float*    __restrict__ dst_f) {
  const int lane  = threadIdx.x;
  const int mtile = blockIdx.y * blockDim.y + threadIdx.y;
  const int nloc  = lane & 15;
  const int mbase = mtile << 4;
  const int nbase = blockIdx.x << 5;          // 32 columns per wave

  const _Float16* arow  = X + (size_t)(mbase + nloc) * C_;
  const _Float16* brow0 = W + (size_t)(nbase + nloc) * C_;
  const _Float16* brow1 = W + (size_t)(nbase + 16 + nloc) * C_;

  v8f acc0 = {}, acc1 = {};
#pragma unroll
  for (int kk = 0; kk < 8; ++kk) {
    const int kb = kk << 5;
    if (kk < 7) {  // CDNA5 global_prefetch_b8 for the next K slice
      __builtin_prefetch(arow + kb + 32, 0, 0);
    }
    v16h a = loadA(arow + kb, lane);
    acc0 = wmma16x16x32(a, loadB(brow0 + kb, lane), acc0);
    acc1 = wmma16x16x32(a, loadB(brow1 + kb, lane), acc1);
  }

  const int mo = (lane >> 4) << 3;
#pragma unroll
  for (int half = 0; half < 2; ++half) {
    const int dtot = nbase + half * 16 + nloc;
    const v8f& acc = half ? acc1 : acc0;
#pragma unroll
    for (int r = 0; r < 8; ++r) {
      const int m = mbase + r + mo;
      const int s = m >> 8, qi = m & 255;
      const float y = acc[r];
      if (MODE == 0 || MODE == 1) {
        const int h = dtot >> 5, d = dtot & 31;
        dst_h[(((size_t)(s * H_ + h) * Q_) + qi) * DH_ + d] = (_Float16)y;
      } else if (MODE == 2) {
        const int h = dtot >> 5, d = dtot & 31;
        dst_h[(((size_t)(s * H_ + h) * DH_) + d) * Q_ + qi] = (_Float16)y;
      } else {
        dst_f[(size_t)m * TOT_ + dtot] =
            1.0f / (1.0f + __expf(-(y + bg[dtot])));
      }
    }
  }
}

// --------------------------------------------------------------- attention ---
// One block per (s,h). 8 waves; wave w owns row strips w and w+8 (16 rows each).
// Online softmax: one 32-column score slab live at a time; the running max is
// row-wide (permlane16 butterfly), the running sum stays lane-local and is
// reduced once per strip at the end.
__global__ __launch_bounds__(256) void attn_kernel(
    const _Float16* __restrict__ qh_all,  // [S*H, Q, DH]
    const _Float16* __restrict__ kh_all,  // [S*H, Q, DH]
    const _Float16* __restrict__ vt_all,  // [S*H, DH, Q]
    const float* __restrict__ bias_mask,  // [S, Q]
    const float* __restrict__ bias_pair,  // [H, Q, Q]
    const float* __restrict__ g,          // [M, TOT]
    _Float16* __restrict__ o_buf) {       // [M, TOT]
  __shared__ __align__(16) _Float16 pbuf[8][16 * 32];  // per-wave prob strip

  const int lane = threadIdx.x;
  const int w    = threadIdx.y;
  const int sh   = blockIdx.x;
  const int s = sh >> 3, h = sh & 7;

  const _Float16* qh = qh_all + (size_t)sh * Q_ * DH_;
  const _Float16* kh = kh_all + (size_t)sh * Q_ * DH_;
  const _Float16* vt = vt_all + (size_t)sh * DH_ * Q_;
  const float* bm = bias_mask + (size_t)s * Q_;
  const float* bp = bias_pair + (size_t)h * Q_ * Q_;

  const int nloc = lane & 15;
  const int mo   = (lane >> 4) << 3;
  _Float16* pl = &pbuf[w][0];

  for (int it = 0; it < 2; ++it) {
    const int mbase = (w + 8 * it) << 4;

    // Q fragment: DH=32 == one full WMMA K dimension.
    v16h aq = loadA(qh + (size_t)(mbase + nloc) * DH_, lane);

    v8f o0 = {}, o1 = {};
    float rmax[8], rsum[8];
#pragma unroll
    for (int r = 0; r < 8; ++r) { rmax[r] = -3.0e38f; rsum[r] = 0.0f; }

    for (int kk = 0; kk < 8; ++kk) {         // 32 key-columns per step
      const int cb = kk << 5;
      // scores for two 16-column tiles
      v8f z = {};
      v8f s0 = wmma16x16x32(aq, loadB(kh + (size_t)(cb + nloc) * DH_, lane), z);
      v8f s1 = wmma16x16x32(aq, loadB(kh + (size_t)(cb + 16 + nloc) * DH_, lane), z);

      const int col0 = cb + nloc, col1 = col0 + 16;
      const float bm0 = bm[col0], bm1 = bm[col1];
#pragma unroll
      for (int r = 0; r < 8; ++r) {
        const size_t row = (size_t)(mbase + r + mo) * Q_;
        s0[r] += bm0 + bp[row + col0];
        s1[r] += bm1 + bp[row + col1];
      }

      // online softmax update (row == (VGPR slot r) x (16-lane half)).
      // Row-wide max (uniform across the 16 lanes) -> rescale factor is
      // row-uniform, so the per-lane partial sums stay mutually consistent.
#pragma unroll
      for (int r = 0; r < 8; ++r) {
        const float mloc = half_reduce_max(fmaxf(s0[r], s1[r]));
        const float mnew = fmaxf(rmax[r], mloc);
        const float scale = __expf(rmax[r] - mnew);
        rmax[r] = mnew;
        o0[r] *= scale;
        o1[r] *= scale;
        const float p0 = __expf(s0[r] - mnew);
        const float p1 = __expf(s1[r] - mnew);
        rsum[r] = rsum[r] * scale + (p0 + p1);   // lane-local partial
        pl[(r + mo) * 32 + nloc]      = (_Float16)p0;
        pl[(r + mo) * 32 + 16 + nloc] = (_Float16)p1;
      }

      // o += P_slab @ V_slab  (re-layout C-format -> A-format via LDS strip)
      v16h pa = loadA(pl + nloc * 32, lane);                      // ds_load_b128
      v16h b0 = loadB(vt + (size_t)nloc * Q_ + cb, lane);
      v16h b1 = loadB(vt + (size_t)(16 + nloc) * Q_ + cb, lane);
      o0 = wmma16x16x32(pa, b0, o0);
      o1 = wmma16x16x32(pa, b1, o1);
    }

    // one cross-lane sum reduction per strip; normalize, gate, emit f16
#pragma unroll
    for (int r = 0; r < 8; ++r) {
      const float inv = 1.0f / half_reduce_sum(rsum[r]);
      const int qi = mbase + r + mo;
      const size_t row = (size_t)(s * Q_ + qi) * TOT_ + h * DH_;
      o_buf[row + nloc]      = (_Float16)(o0[r] * inv * g[row + nloc]);
      o_buf[row + 16 + nloc] = (_Float16)(o1[r] * inv * g[row + 16 + nloc]);
    }
  }
}

// -------------------------------------------------------- output projection ---
__global__ __launch_bounds__(256) void out_kernel(
    const _Float16* __restrict__ O,   // [M, TOT] f16
    const _Float16* __restrict__ Wo,  // [C, TOT] f16
    const float* __restrict__ bo,     // [C]
    float* __restrict__ out) {        // [M, C] f32
  const int lane  = threadIdx.x;
  const int mtile = blockIdx.y * blockDim.y + threadIdx.y;
  const int nloc  = lane & 15;
  const int mbase = mtile << 4;
  const int nbase = blockIdx.x << 5;

  const _Float16* arow  = O  + (size_t)(mbase + nloc) * TOT_;
  const _Float16* brow0 = Wo + (size_t)(nbase + nloc) * TOT_;
  const _Float16* brow1 = Wo + (size_t)(nbase + 16 + nloc) * TOT_;

  v8f acc0 = {}, acc1 = {};
#pragma unroll
  for (int kk = 0; kk < 8; ++kk) {
    const int kb = kk << 5;
    v16h a = loadA(arow + kb, lane);
    acc0 = wmma16x16x32(a, loadB(brow0 + kb, lane), acc0);
    acc1 = wmma16x16x32(a, loadB(brow1 + kb, lane), acc1);
  }

  const int mo = (lane >> 4) << 3;
#pragma unroll
  for (int half = 0; half < 2; ++half) {
    const int cc = nbase + half * 16 + nloc;
    const float bb = bo[cc];
    const v8f& acc = half ? acc1 : acc0;
#pragma unroll
    for (int r = 0; r < 8; ++r)
      out[(size_t)(mbase + r + mo) * C_ + cc] = acc[r] + bb;
  }
}

// ------------------------------------------------------------------- launch ---
extern "C" void kernel_launch(void* const* d_in, const int* in_sizes, int n_in,
                              void* d_out, int out_size, void* d_ws,
                              size_t ws_size, hipStream_t stream) {
  (void)in_sizes; (void)n_in; (void)out_size; (void)ws_size;

  const float* q_x       = (const float*)d_in[0];
  const float* kv_x      = (const float*)d_in[1];
  const float* bias_mask = (const float*)d_in[2];
  const float* bias_pair = (const float*)d_in[3];
  const float* Wq        = (const float*)d_in[4];
  const float* Wk        = (const float*)d_in[5];
  const float* Wv        = (const float*)d_in[6];
  const float* Wg        = (const float*)d_in[7];
  const float* bg        = (const float*)d_in[8];
  const float* Wo        = (const float*)d_in[9];
  const float* bo        = (const float*)d_in[10];

  char* ws = (char*)d_ws;
  size_t off = 0;
  auto take = [&](size_t bytes) -> char* {
    char* p = ws + off;
    off = (off + bytes + 255) & ~(size_t)255;
    return p;
  };
  _Float16* Xq   = (_Float16*)take((size_t)M_ * C_ * 2);
  _Float16* Xkv  = (_Float16*)take((size_t)M_ * C_ * 2);
  _Float16* Wq_h = (_Float16*)take((size_t)TOT_ * C_ * 2);
  _Float16* Wk_h = (_Float16*)take((size_t)TOT_ * C_ * 2);
  _Float16* Wv_h = (_Float16*)take((size_t)TOT_ * C_ * 2);
  _Float16* Wg_h = (_Float16*)take((size_t)TOT_ * C_ * 2);
  _Float16* Wo_h = (_Float16*)take((size_t)TOT_ * C_ * 2);
  _Float16* qhd  = (_Float16*)take((size_t)S_ * H_ * Q_ * DH_ * 2);
  _Float16* khd  = (_Float16*)take((size_t)S_ * H_ * Q_ * DH_ * 2);
  _Float16* vtd  = (_Float16*)take((size_t)S_ * H_ * DH_ * Q_ * 2);
  float*    gbuf = (float*)take((size_t)M_ * TOT_ * 4);
  _Float16* obuf = Xkv;  // kv f16 copy is dead once k/v projections finish

  // fp32 -> f16 conversions
  cvt_f32_f16<<<512, 256, 0, stream>>>(q_x,  Xq,  M_ * C_);
  cvt_f32_f16<<<512, 256, 0, stream>>>(kv_x, Xkv, M_ * C_);
  cvt_f32_f16<<<64, 256, 0, stream>>>(Wq, Wq_h, TOT_ * C_);
  cvt_f32_f16<<<64, 256, 0, stream>>>(Wk, Wk_h, TOT_ * C_);
  cvt_f32_f16<<<64, 256, 0, stream>>>(Wv, Wv_h, TOT_ * C_);
  cvt_f32_f16<<<64, 256, 0, stream>>>(Wg, Wg_h, TOT_ * C_);
  cvt_f32_f16<<<64, 256, 0, stream>>>(Wo, Wo_h, TOT_ * C_);

  const dim3 blk(32, 8);                       // 8 wave32 waves
  const dim3 grd(TOT_ / 32, M_ / 16 / 8);      // (8, 256): 16x32 tile per wave

  proj_kernel<0><<<grd, blk, 0, stream>>>(Xq,  Wq_h, nullptr, qhd, nullptr);
  proj_kernel<1><<<grd, blk, 0, stream>>>(Xkv, Wk_h, nullptr, khd, nullptr);
  proj_kernel<2><<<grd, blk, 0, stream>>>(Xkv, Wv_h, nullptr, vtd, nullptr);
  proj_kernel<3><<<grd, blk, 0, stream>>>(Xq,  Wg_h, bg, nullptr, gbuf);

  attn_kernel<<<S_ * H_, blk, 0, stream>>>(qhd, khd, vtd, bias_mask, bias_pair,
                                           gbuf, obuf);

  out_kernel<<<grd, blk, 0, stream>>>(obuf, Wo_h, bo, (float*)d_out);
}
